// MultiSimilarityLoss_46540265619750
// MI455X (gfx1250) — matrix operand order repaired and checked by reading, not compiled, not run on hardware
//
#include <hip/hip_runtime.h>
#include <cmath>

// ---------------------------------------------------------------------------
// MultiSimilarityLoss for MI455X (gfx1250, wave32, WMMA)
//   feats: [8192, 512] f32, labels: [8192] i32  ->  (loss, avg_ap, avg_an)
//
// Pipeline:
//   K1: L2-normalize rows, split fp32 -> bf16 hi + bf16 lo (ws)
//   K2: fused 2-sweep row-block kernel; sim tiles via v_wmma_f32_16x16x32_bf16
//       (hi*hi + hi*lo + lo*hi  ~ fp32-accurate sim)
//   K3: deterministic tree reduction of per-row results -> 3 scalars
// ---------------------------------------------------------------------------

typedef __attribute__((ext_vector_type(16))) __bf16 v16bf;
typedef __attribute__((ext_vector_type(8)))  __bf16 v8bf;
typedef __attribute__((ext_vector_type(8)))  float  v8f;

#define NB        8192
#define DD        512
#define KCHUNKS   (DD / 32)     // 16 K-chunks of 32
#define NT        8             // 16-col tiles per column group
#define CGCOLS    (16 * NT)     // 128 columns per group
#define NCG       (NB / CGCOLS) // 64 column groups
#define ROWS_PER_WG 64          // 4 waves x 16 rows
#define WG_THREADS  128

#define MS_THRESH 0.5f
#define MS_MARGIN 0.1f
#define MS_SPOS   2.0f
#define MS_SNEG   40.0f

// ---- 16-lane (half-wave) butterfly reductions (wave32) ---------------------
__device__ __forceinline__ float red_sum16(float x) {
  x += __shfl_xor(x, 1, 16); x += __shfl_xor(x, 2, 16);
  x += __shfl_xor(x, 4, 16); x += __shfl_xor(x, 8, 16);
  return x;
}
__device__ __forceinline__ float red_min16(float x) {
  x = fminf(x, __shfl_xor(x, 1, 16)); x = fminf(x, __shfl_xor(x, 2, 16));
  x = fminf(x, __shfl_xor(x, 4, 16)); x = fminf(x, __shfl_xor(x, 8, 16));
  return x;
}
__device__ __forceinline__ float red_max16(float x) {
  x = fmaxf(x, __shfl_xor(x, 1, 16)); x = fmaxf(x, __shfl_xor(x, 2, 16));
  x = fmaxf(x, __shfl_xor(x, 4, 16)); x = fmaxf(x, __shfl_xor(x, 8, 16));
  return x;
}

// ---- A-fragment (16x32 bf16), contiguous-K layout (B uses the same order):
//   lane L: row M = L%16 ; g = L/16
//   elems 0..7  <- K = g*16 + 0..7       (contiguous 16B)
//   elems 8..15 <- K = g*16 + 8 + 0..7   (contiguous 16B)
__device__ __forceinline__ v16bf load_a(const __bf16* p, int g) {
  union { v16bf v; v8bf h[2]; } u;
  u.h[0] = *reinterpret_cast<const v8bf*>(p + g * 16);
  u.h[1] = *reinterpret_cast<const v8bf*>(p + g * 16 + 8);
  return u.v;
}

// ---- column sweep over the whole 8192-col dimension; consumer(sim, v, labC, c)
template <typename Consumer>
__device__ __forceinline__ void sweep_all_cols(
    const __bf16* __restrict__ fh, const __bf16* __restrict__ fl,
    const __bf16* arh, const __bf16* arl,
    const int* __restrict__ labels,
    int g, int ln, Consumer&& consume)
{
  for (int cgi = 0; cgi < NCG; ++cgi) {
    const int cbase = cgi * CGCOLS;
    int labC[NT];
#pragma unroll
    for (int t = 0; t < NT; ++t) labC[t] = labels[cbase + t * 16 + ln];

    v8f acc[NT] = {};
    for (int kc = 0; kc < KCHUNKS; ++kc) {
      const int kb = kc * 32;
      const v16bf ah = load_a(arh + kb, g);
      const v16bf al = load_a(arl + kb, g);
#pragma unroll
      for (int t = 0; t < NT; ++t) {
        // B-fragment (32x16 bf16): lane L: col N = L%16, K = g*16 .. g*16+15
        // = 16 contiguous bf16 of row (col) in row-major f -> one 32B load
        const size_t coff = (size_t)(cbase + t * 16 + ln) * DD + kb + g * 16;
        const v16bf bh = *reinterpret_cast<const v16bf*>(fh + coff);
        const v16bf bl = *reinterpret_cast<const v16bf*>(fl + coff);
        acc[t] = __builtin_amdgcn_wmma_f32_16x16x32_bf16(
            false, ah, false, bh, (short)0, acc[t], false, false);
        acc[t] = __builtin_amdgcn_wmma_f32_16x16x32_bf16(
            false, ah, false, bl, (short)0, acc[t], false, false);
        acc[t] = __builtin_amdgcn_wmma_f32_16x16x32_bf16(
            false, al, false, bh, (short)0, acc[t], false, false);
      }
    }
#pragma unroll
    for (int t = 0; t < NT; ++t) {
      const int c = cbase + t * 16 + ln;
#pragma unroll
      for (int v = 0; v < 8; ++v) consume(acc[t][v], v, labC[t], c);
    }
    __syncthreads();  // loosely phase-align waves for WGP$ reuse of B stream
  }
}

// ---------------------------------------------------------------------------
// K1: row L2-normalize + bf16 hi/lo split. One wave per row.
// ---------------------------------------------------------------------------
__global__ __launch_bounds__(256)
void ms_normalize_split(const float* __restrict__ f,
                        __bf16* __restrict__ fh, __bf16* __restrict__ fl)
{
  const int lane = threadIdx.x & 31;
  const int row  = (blockIdx.x * blockDim.x + threadIdx.x) >> 5;
  if (row >= NB) return;
  const float* r = f + (size_t)row * DD;
  float ss = 0.f;
#pragma unroll 4
  for (int i = lane; i < DD; i += 32) { const float x = r[i]; ss += x * x; }
  ss += __shfl_xor(ss, 1, 32); ss += __shfl_xor(ss, 2, 32);
  ss += __shfl_xor(ss, 4, 32); ss += __shfl_xor(ss, 8, 32);
  ss += __shfl_xor(ss, 16, 32);
  const float inv = 1.0f / sqrtf(ss);
#pragma unroll 4
  for (int i = lane; i < DD; i += 32) {
    const float x = r[i] * inv;
    const __bf16 h = (__bf16)x;
    const float  l = x - (float)h;
    fh[(size_t)row * DD + i] = h;
    fl[(size_t)row * DD + i] = (__bf16)l;
  }
}

// ---------------------------------------------------------------------------
// K2: fused two-sweep multi-similarity per 64-row block
// ---------------------------------------------------------------------------
__global__ __launch_bounds__(WG_THREADS)
void ms_main(const __bf16* __restrict__ fh, const __bf16* __restrict__ fl,
             const int* __restrict__ labels,
             float* __restrict__ row_loss, float* __restrict__ row_map,
             float* __restrict__ row_man)
{
  const int tid  = threadIdx.x;
  const int lane = tid & 31;
  const int wave = tid >> 5;
  const int g    = lane >> 4;   // half-wave id
  const int ln   = lane & 15;
  const int rA   = blockIdx.x * ROWS_PER_WG + wave * 16;

  // A row for this lane (A layout: M = lane%16)
  const __bf16* arh = fh + (size_t)(rA + ln) * DD;
  const __bf16* arl = fl + (size_t)(rA + ln) * DD;

  // D-tile layout: element (v, lane) -> row rA + v + 8*g, col = cbase + ln
  int labR[8];
#pragma unroll
  for (int v = 0; v < 8; ++v) labR[v] = labels[rA + v + 8 * g];

  // ---------------- sweep 1: counts / sums / min_pos / max_neg -------------
  float pcnt[8], psum[8], pmin[8], ncnt[8], nsum[8], nmax[8];
#pragma unroll
  for (int v = 0; v < 8; ++v) {
    pcnt[v] = 0.f; psum[v] = 0.f; ncnt[v] = 0.f; nsum[v] = 0.f;
    pmin[v] = __builtin_inff(); nmax[v] = -__builtin_inff();
  }

  sweep_all_cols(fh, fl, arh, arl, labels, g, ln,
    [&](float sim, int v, int labc, int c) {
      const int  r    = rA + v + 8 * g;
      const bool same = (labR[v] == labc);
      const bool pos  = same && (r != c);   // sim<1-eps excludes only the diag
      const bool neg  = !same;
      pcnt[v] += pos ? 1.f : 0.f;
      psum[v] += pos ? sim : 0.f;
      pmin[v]  = pos ? fminf(pmin[v], sim) : pmin[v];
      ncnt[v] += neg ? 1.f : 0.f;
      nsum[v] += neg ? sim : 0.f;
      nmax[v]  = neg ? fmaxf(nmax[v], sim) : nmax[v];
    });

  float valid1[8], mp[8], mn[8];
#pragma unroll
  for (int v = 0; v < 8; ++v) {
    pcnt[v] = red_sum16(pcnt[v]); psum[v] = red_sum16(psum[v]);
    ncnt[v] = red_sum16(ncnt[v]); nsum[v] = red_sum16(nsum[v]);
    pmin[v] = red_min16(pmin[v]); nmax[v] = red_max16(nmax[v]);
    valid1[v] = (pcnt[v] >= 1.f && ncnt[v] >= 1.f) ? 1.f : 0.f;
    mp[v] = psum[v] / fmaxf(pcnt[v], 1.f);
    mn[v] = nsum[v] / fmaxf(ncnt[v], 1.f);
  }
  // after butterfly, every lane holds its own row's pmin/nmax -> reuse directly

  // ---------------- sweep 2: hard-mined exp sums ----------------------------
  float pexp[8], nexp[8], pscnt[8], nscnt[8];
#pragma unroll
  for (int v = 0; v < 8; ++v) { pexp[v] = nexp[v] = pscnt[v] = nscnt[v] = 0.f; }

  sweep_all_cols(fh, fl, arh, arl, labels, g, ln,
    [&](float sim, int v, int labc, int c) {
      const int  r    = rA + v + 8 * g;
      const bool same = (labR[v] == labc);
      const bool pos  = same && (r != c);
      const bool neg  = !same;
      const bool nsel = neg && (sim + MS_MARGIN > pmin[v]);
      const bool psel = pos && (sim - MS_MARGIN < nmax[v]);
      nexp[v]  += nsel ? __expf(MS_SNEG * (sim - MS_THRESH)) : 0.f;
      nscnt[v] += nsel ? 1.f : 0.f;
      pexp[v]  += psel ? __expf(-MS_SPOS * (sim - MS_THRESH)) : 0.f;
      pscnt[v] += psel ? 1.f : 0.f;
    });

#pragma unroll
  for (int v = 0; v < 8; ++v) {
    pexp[v]  = red_sum16(pexp[v]);  nexp[v]  = red_sum16(nexp[v]);
    pscnt[v] = red_sum16(pscnt[v]); nscnt[v] = red_sum16(nscnt[v]);
  }

  if (ln == 0) {
#pragma unroll
    for (int v = 0; v < 8; ++v) {
      const int   r      = rA + v + 8 * g;
      const float valid2 = (pscnt[v] >= 1.f && nscnt[v] >= 1.f) ? 1.f : 0.f;
      const float valid  = valid1[v] * valid2;
      const float rl = (1.f / MS_SPOS) * log1pf(pexp[v]) +
                       (1.f / MS_SNEG) * log1pf(nexp[v]);
      row_loss[r] = valid * rl;
      row_map[r]  = valid1[v] * mp[v];
      row_man[r]  = valid1[v] * mn[v];
    }
  }
}

// ---------------------------------------------------------------------------
// K3: deterministic reduction of per-row results -> 3 scalars
// ---------------------------------------------------------------------------
__global__ __launch_bounds__(256)
void ms_finalize(const float* __restrict__ rl, const float* __restrict__ rp,
                 const float* __restrict__ rn, float* __restrict__ out)
{
  __shared__ float sl[256], sp[256], sn[256];
  const int t = threadIdx.x;
  float a = 0.f, b = 0.f, c = 0.f;
  for (int i = t; i < NB; i += 256) { a += rl[i]; b += rp[i]; c += rn[i]; }
  sl[t] = a; sp[t] = b; sn[t] = c;
  __syncthreads();
  for (int s = 128; s > 0; s >>= 1) {
    if (t < s) { sl[t] += sl[t + s]; sp[t] += sp[t + s]; sn[t] += sn[t + s]; }
    __syncthreads();
  }
  if (t == 0) {
    const float ib = 1.0f / (float)NB;
    out[0] = sl[0] * ib;   // loss
    out[1] = sp[0] * ib;   // avg_ap
    out[2] = sn[0] * ib;   // avg_an
  }
}

// ---------------------------------------------------------------------------
extern "C" void kernel_launch(void* const* d_in, const int* in_sizes, int n_in,
                              void* d_out, int out_size, void* d_ws, size_t ws_size,
                              hipStream_t stream)
{
  const float* feats  = (const float*)d_in[0];
  const int*   labels = (const int*)d_in[1];

  // workspace layout: fh[8M] bf16 | fl[8M] bf16 | 3 x 8192 f32  (~16.9 MB)
  __bf16* fh = (__bf16*)d_ws;
  __bf16* fl = fh + (size_t)NB * DD;
  float*  rl = (float*)(fl + (size_t)NB * DD);
  float*  rp = rl + NB;
  float*  rn = rp + NB;

  ms_normalize_split<<<dim3(NB / 8), dim3(256), 0, stream>>>(feats, fh, fl);
  ms_main<<<dim3(NB / ROWS_PER_WG), dim3(WG_THREADS), 0, stream>>>(
      fh, fl, labels, rl, rp, rn);
  ms_finalize<<<dim3(1), dim3(256), 0, stream>>>(rl, rp, rn, (float*)d_out);
}
